// GNNPlus_472446402724
// MI455X (gfx1250) — matrix operand order repaired and checked by reading, not compiled, not run on hardware
//
#include <hip/hip_runtime.h>
#include <hip/hip_bf16.h>
#include <stdint.h>

#define NN 20000      // nodes
#define NE 320000     // edges
#define DIN 64        // node_in
#define EIN 16        // edge_in
#define HD 256        // hidden
#define NG 64         // graphs

typedef __attribute__((ext_vector_type(16))) __bf16 v16bf;
typedef __attribute__((ext_vector_type(8)))  float  v8f;
typedef __attribute__((ext_vector_type(8)))  unsigned short v8us;
typedef __attribute__((ext_vector_type(16))) unsigned short v16us;

// native hardware convert (v_cvt_*bf16_f32), RNE
__device__ __forceinline__ unsigned short f2bf(float f) {
    __bf16 h = (__bf16)f;
    return __builtin_bit_cast(unsigned short, h);
}

__device__ __forceinline__ v8f vzero8() {
    v8f c;
#pragma unroll
    for (int i = 0; i < 8; ++i) c[i] = 0.0f;
    return c;
}

__device__ __forceinline__ v16bf frag_cast(v8us lo, v8us hi) {
    v16us u;
#pragma unroll
    for (int i = 0; i < 8; ++i) { u[i] = lo[i]; u[i + 8] = hi[i]; }
    return __builtin_bit_cast(v16bf, u);
}

// A fragment, 16x32 bf16, row-major source with leading dim ld.
// Lane<16: K k0..k0+7 and k0+16..k0+23 ; Lane>=16: +8 on both runs.
__device__ __forceinline__ v16bf load_a(const unsigned short* base, int ld, int row0, int k0) {
    int lane = threadIdx.x & 31;
    int row = row0 + (lane & 15);
    int k = k0 + ((lane >> 4) << 3);
    const unsigned short* p = base + (size_t)row * ld + k;
    v8us lo = *(const v8us*)p;
    v8us hi = *(const v8us*)(p + 16);
    return frag_cast(lo, hi);
}

// Same but with row clamped (tail guard for node kernels)
__device__ __forceinline__ v16bf load_a_cl(const unsigned short* base, int ld, int row0, int k0, int rmax) {
    int lane = threadIdx.x & 31;
    int row = row0 + (lane & 15);
    row = row < rmax ? row : rmax - 1;
    int k = k0 + ((lane >> 4) << 3);
    const unsigned short* p = base + (size_t)row * ld + k;
    v8us lo = *(const v8us*)p;
    v8us hi = *(const v8us*)(p + 16);
    return frag_cast(lo, hi);
}

// A fragment when only K=16 is valid (edge_attr, ld=16): upper half zero.
__device__ __forceinline__ v16bf load_a_k16(const unsigned short* base, int row0) {
    int lane = threadIdx.x & 31;
    int row = row0 + (lane & 15);
    int k = (lane >> 4) << 3;
    v8us lo = *(const v8us*)(base + (size_t)row * 16 + k);
    v8us hi;
#pragma unroll
    for (int i = 0; i < 8; ++i) hi[i] = 0;
    return frag_cast(lo, hi);
}

// B fragment, 32x16 bf16 from transposed weights wt[Ncols x K] (per-column contiguous K).
// Lane = column; Lane<16: K k0..k0+7 and k0+16..k0+23 ; Lane>=16: +8 on both runs.
__device__ __forceinline__ v16bf load_b(const unsigned short* wt, int K, int n0, int k0) {
    int lane = threadIdx.x & 31;
    int col = n0 + (lane & 15);
    int k = k0 + ((lane >> 4) << 3);
    const unsigned short* p = wt + (size_t)col * K + k;
    v8us lo = *(const v8us*)p;
    v8us hi = *(const v8us*)(p + 16);
    return frag_cast(lo, hi);
}

__device__ __forceinline__ v8f wmma_bf16(v16bf a, v16bf b, v8f c) {
    return __builtin_amdgcn_wmma_f32_16x16x32_bf16(false, a, false, b, (short)0, c, false, false);
}

// gfx1250 async global->LDS copy, 16 bytes per enabled lane (ASYNCcnt tracked).
// LDS destination address = low 32 bits of the generic shared-memory pointer
__device__ __forceinline__ void async_copy_b128(uint32_t lds_addr, const void* gaddr) {
    asm volatile("global_load_async_to_lds_b128 %0, %1, off"
                 :: "v"(lds_addr), "v"(gaddr) : "memory");
}
__device__ __forceinline__ void wait_async0() {
    asm volatile("s_wait_asynccnt 0" ::: "memory");
}

// ---------------- utility kernels ----------------

__global__ void k_zero_f32(float* p, int n) {
    int i = blockIdx.x * blockDim.x + threadIdx.x;
    if (i < n) p[i] = 0.0f;
}

__global__ void k_f32_to_bf16(const float* src, unsigned short* dst, int n) {
    int i = blockIdx.x * blockDim.x + threadIdx.x;
    if (i < n) dst[i] = f2bf(src[i]);
}

__global__ void k_add_bf16(const float* a, const float* b, unsigned short* dst, int n) {
    int i = blockIdx.x * blockDim.x + threadIdx.x;
    if (i < n) dst[i] = f2bf(a[i] + b[i]);
}

// transpose+convert weight [K x Nc] (row-major) -> bf16 [Nc x Kpad], zero-padded in K
__global__ void k_wt_bf16(const float* w, unsigned short* wt, int K, int Nc, int Kpad) {
    int i = blockIdx.x * blockDim.x + threadIdx.x;
    if (i >= Nc * Kpad) return;
    int n = i / Kpad, k = i - n * Kpad;
    wt[i] = (k < K) ? f2bf(w[(size_t)k * Nc + n]) : (unsigned short)0;
}

__global__ void k_counts(const int* batch, float* cnt) {
    int i = blockIdx.x * blockDim.x + threadIdx.x;
    if (i < NN) atomicAdd(&cnt[batch[i]], 1.0f);
}

// ---------------- edge MLP: ea = relu(edge_attr@ew1+b1)@ew2+b2, stored bf16 ----------------

__global__ __launch_bounds__(256) void k_edge_mlp(
    const unsigned short* __restrict__ eattr,   // [E x 16] bf16
    const unsigned short* __restrict__ ew1t,    // [256 x 32] bf16 (K padded 16->32)
    const float* __restrict__ eb1,
    const unsigned short* __restrict__ ew2t,    // [256 x 256] bf16
    const float* __restrict__ eb2,
    unsigned short* __restrict__ ea)            // [E x 256] bf16
{
    const int HS = 264;                          // LDS row stride (ushorts), 16B aligned
    __shared__ unsigned short hid[64 * 264];
    int wave = threadIdx.x >> 5, lane = threadIdx.x & 31;
    int e0 = blockIdx.x * 64;
    int cl = lane & 15, mb = (lane >> 4) << 3;

    // layer 1: 4 row-tiles x 16 n-tiles, K=16 (one padded WMMA each)
    for (int t = wave; t < 64; t += 8) {
        int rt = t >> 4, nt = t & 15;
        v16bf a = load_a_k16(eattr, e0 + rt * 16);
        v16bf b = load_b(ew1t, 32, nt * 16, 0);
        v8f c = vzero8();
        c = wmma_bf16(a, b, c);
        int col = nt * 16 + cl;
        float bias = eb1[col];
#pragma unroll
        for (int r = 0; r < 8; ++r) {
            float v = c[r] + bias;
            v = v > 0.0f ? v : 0.0f;
            hid[(rt * 16 + mb + r) * HS + col] = f2bf(v);
        }
    }
    __syncthreads();
    // layer 2: K=256
    for (int t = wave; t < 64; t += 8) {
        int rt = t >> 4, nt = t & 15;
        int n0 = nt * 16;
        v8f c = vzero8();
#pragma unroll
        for (int kk = 0; kk < 256; kk += 32) {
            v16bf a = load_a(hid, HS, rt * 16, kk);
            v16bf b = load_b(ew2t, 256, n0, kk);
            c = wmma_bf16(a, b, c);
        }
        int col = n0 + cl;
        float bias = eb2[col];
#pragma unroll
        for (int r = 0; r < 8; ++r) {
            float v = c[r] + bias;
            ea[(size_t)(e0 + rt * 16 + mb + r) * 256 + col] = f2bf(v);
        }
    }
}

// ---------------- fused edge projection + message + scatter ----------------
// proj = ea @ wt + bias ; msg = relu(xf[src] + proj) ; aggr[dst] += msg
// Projection weights [D x 256] staged once per block into LDS via async DMA.
template <int D>
__global__ __launch_bounds__(256) void k_conv_msg(
    const unsigned short* __restrict__ ea,    // [E x 256] bf16
    const unsigned short* __restrict__ wt,    // [D x 256] bf16
    const float* __restrict__ bias,           // [D]
    const float* __restrict__ xf,             // [N x D] gather source (f32)
    const int* __restrict__ srcIdx,
    const int* __restrict__ dstIdx,
    float* __restrict__ aggr)                 // [N x D]
{
    __shared__ unsigned short wlds[D * 256];  // D*512 bytes (32KB or 128KB)
    int wave = threadIdx.x >> 5, lane = threadIdx.x & 31;
    int e0 = (blockIdx.x * 8 + wave) * 16;    // grid exactly covers NE
    int cl = lane & 15, mb = (lane >> 4) << 3;

    // async-stage projection weights into LDS (16B per lane per issue)
    {
        uint32_t lbase = (uint32_t)(uintptr_t)(void*)wlds;
        const char* gbase = (const char*)wt;
#pragma unroll
        for (int it = 0; it < D / 8; ++it) {
            uint32_t off = (uint32_t)(it * 256 + threadIdx.x) * 16u;
            async_copy_b128(lbase + off, gbase + off);
        }
        wait_async0();
    }

    __builtin_prefetch(ea + (size_t)(e0 + 128) * 256, 0, 0);

    v16bf afr[8];
#pragma unroll
    for (int i = 0; i < 8; ++i) afr[i] = load_a(ea, 256, e0, i * 32);

    int sidx[8], didx[8];
#pragma unroll
    for (int r = 0; r < 8; ++r) {
        int e = e0 + mb + r;
        sidx[r] = srcIdx[e];
        didx[r] = dstIdx[e];
    }

    __syncthreads();   // all waves' async weight DMA complete

    for (int nt = 0; nt < D / 16; ++nt) {
        int n0 = nt * 16;
        v8f c = vzero8();
#pragma unroll
        for (int i = 0; i < 8; ++i) {
            v16bf b = load_b(wlds, 256, n0, i * 32);
            c = wmma_bf16(afr[i], b, c);
        }
        int col = n0 + cl;
        float bv = bias[col];
#pragma unroll
        for (int r = 0; r < 8; ++r) {
            float v = c[r] + bv + xf[(size_t)sidx[r] * D + col];
            v = v > 0.0f ? v : 0.0f;
            atomicAdd(&aggr[(size_t)didx[r] * D + col], v);
        }
    }
}

// ---------------- node MLP: out = relu( relu(t@w1+b1)@w2+b2 ) ----------------
template <int K1, bool STORE_F32, bool POOL>
__global__ __launch_bounds__(256) void k_node_mlp(
    const unsigned short* __restrict__ tin,   // [N x K1] bf16
    const unsigned short* __restrict__ w1t,   // [256 x K1] bf16
    const float* __restrict__ b1,
    const unsigned short* __restrict__ w2t,   // [256 x 256] bf16
    const float* __restrict__ b2,
    float* __restrict__ outf,                 // [N x 256] (STORE_F32)
    const int* __restrict__ batch,            // (POOL)
    float* __restrict__ pool)                 // [G x 256] (POOL)
{
    const int HS = 264;
    __shared__ unsigned short hid[64 * 264];
    int wave = threadIdx.x >> 5, lane = threadIdx.x & 31;
    int r0 = blockIdx.x * 64;
    int cl = lane & 15, mb = (lane >> 4) << 3;

    for (int t = wave; t < 64; t += 8) {
        int rt = t >> 4, nt = t & 15;
        int n0 = nt * 16;
        v8f c = vzero8();
#pragma unroll
        for (int i = 0; i < K1 / 32; ++i) {
            v16bf a = load_a_cl(tin, K1, r0 + rt * 16, i * 32, NN);
            v16bf b = load_b(w1t, K1, n0, i * 32);
            c = wmma_bf16(a, b, c);
        }
        int col = n0 + cl;
        float bias = b1[col];
#pragma unroll
        for (int r = 0; r < 8; ++r) {
            float v = c[r] + bias;
            v = v > 0.0f ? v : 0.0f;
            hid[(rt * 16 + mb + r) * HS + col] = f2bf(v);
        }
    }
    __syncthreads();
    for (int t = wave; t < 64; t += 8) {
        int rt = t >> 4, nt = t & 15;
        int n0 = nt * 16;
        v8f c = vzero8();
#pragma unroll
        for (int kk = 0; kk < 256; kk += 32) {
            v16bf a = load_a(hid, HS, rt * 16, kk);
            v16bf b = load_b(w2t, 256, n0, kk);
            c = wmma_bf16(a, b, c);
        }
        int col = n0 + cl;
        float bias = b2[col];
#pragma unroll
        for (int r = 0; r < 8; ++r) {
            int row = r0 + rt * 16 + mb + r;
            if (row < NN) {
                float v = c[r] + bias;
                v = v > 0.0f ? v : 0.0f;           // outer ReLU after conv
                if (STORE_F32) outf[(size_t)row * 256 + col] = v;
                if (POOL) atomicAdd(&pool[(size_t)batch[row] * 256 + col], v);
            }
        }
    }
}

// ---------------- readout (tiny, scalar) ----------------
__global__ __launch_bounds__(256) void k_readout(
    const float* __restrict__ pool, const float* __restrict__ cnt,
    const float* __restrict__ l1w, const float* __restrict__ l1b,
    const float* __restrict__ l2w, const float* __restrict__ l2b,
    const float* __restrict__ hsw, const float* __restrict__ hsb,
    const float* __restrict__ hpw, const float* __restrict__ hpb,
    const float* __restrict__ hnw, const float* __restrict__ hnb,
    float* __restrict__ out)
{
    __shared__ float g1[NG * 128];
    __shared__ float g2[NG * 64];
    int tid = threadIdx.x;
    for (int i = tid; i < NG * 128; i += 256) {
        int gr = i >> 7, o = i & 127;
        float cdiv = cnt[gr]; cdiv = cdiv > 1.0f ? cdiv : 1.0f;
        float inv = 1.0f / cdiv;
        float acc = l1b[o];
        for (int k = 0; k < 256; ++k) acc += pool[gr * 256 + k] * inv * l1w[k * 128 + o];
        g1[i] = acc > 0.0f ? acc : 0.0f;
    }
    __syncthreads();
    for (int i = tid; i < NG * 64; i += 256) {
        int gr = i >> 6, o = i & 63;
        float acc = l2b[o];
        for (int k = 0; k < 128; ++k) acc += g1[gr * 128 + k] * l2w[k * 64 + o];
        g2[i] = acc > 0.0f ? acc : 0.0f;
    }
    __syncthreads();
    for (int gr = tid; gr < NG; gr += 256) {
        float s = hsb[0], p = hpb[0], n = hnb[0];
        for (int k = 0; k < 64; ++k) {
            float v = g2[gr * 64 + k];
            s += v * hsw[k]; p += v * hpw[k]; n += v * hnw[k];
        }
        out[gr] = s; out[64 + gr] = p; out[128 + gr] = n;
    }
}

// ---------------- host launcher ----------------
extern "C" void kernel_launch(void* const* d_in, const int* in_sizes, int n_in,
                              void* d_out, int out_size, void* d_ws, size_t ws_size,
                              hipStream_t stream) {
    const float* x         = (const float*)d_in[0];
    const int*   eidx      = (const int*)d_in[1];
    const int*   batch     = (const int*)d_in[2];
    const float* edge_attr = (const float*)d_in[3];
    const float* ew1 = (const float*)d_in[4];  const float* eb1 = (const float*)d_in[5];
    const float* ew2 = (const float*)d_in[6];  const float* eb2 = (const float*)d_in[7];
    const float* le1w = (const float*)d_in[8]; const float* le1b = (const float*)d_in[9];
    const float* le2w = (const float*)d_in[10];const float* le2b = (const float*)d_in[11];
    const float* n1w1 = (const float*)d_in[12];const float* n1b1 = (const float*)d_in[13];
    const float* n1w2 = (const float*)d_in[14];const float* n1b2 = (const float*)d_in[15];
    const float* n2w1 = (const float*)d_in[16];const float* n2b1 = (const float*)d_in[17];
    const float* n2w2 = (const float*)d_in[18];const float* n2b2 = (const float*)d_in[19];
    const float* l1w = (const float*)d_in[20]; const float* l1b = (const float*)d_in[21];
    const float* l2w = (const float*)d_in[22]; const float* l2b = (const float*)d_in[23];
    const float* hsw = (const float*)d_in[24]; const float* hsb = (const float*)d_in[25];
    const float* hpw = (const float*)d_in[26]; const float* hpb = (const float*)d_in[27];
    const float* hnw = (const float*)d_in[28]; const float* hnb = (const float*)d_in[29];
    const int* srcI = eidx;          // edge_index[0]
    const int* dstI = eidx + NE;     // edge_index[1]

    char* ws = (char*)d_ws;
    size_t off = 0;
    auto alloc = [&](size_t bytes) { size_t r = off; off += (bytes + 511) & ~(size_t)511; return r; };

    // zero region first (aggr1, aggr2, pool, cnt contiguous)
    size_t o_aggr1 = alloc((size_t)NN * DIN * 4);
    size_t o_aggr2 = alloc((size_t)NN * HD * 4);
    size_t o_pool  = alloc((size_t)NG * HD * 4);
    size_t o_cnt   = alloc((size_t)NG * 4);
    size_t zero_floats = off / 4;
    size_t o_ea    = alloc((size_t)NE * HD * 2);
    size_t o_eat   = alloc((size_t)NE * EIN * 2);
    size_t o_t1    = alloc((size_t)NN * DIN * 2);
    size_t o_h1    = alloc((size_t)NN * HD * 4);
    size_t o_t2    = alloc((size_t)NN * HD * 2);
    size_t o_ew1t  = alloc((size_t)HD * 32 * 2);
    size_t o_ew2t  = alloc((size_t)HD * HD * 2);
    size_t o_le1t  = alloc((size_t)DIN * HD * 2);
    size_t o_le2t  = alloc((size_t)HD * HD * 2);
    size_t o_n1w1t = alloc((size_t)HD * DIN * 2);
    size_t o_n1w2t = alloc((size_t)HD * HD * 2);
    size_t o_n2w1t = alloc((size_t)HD * HD * 2);
    size_t o_n2w2t = alloc((size_t)HD * HD * 2);

    float* aggr1 = (float*)(ws + o_aggr1);
    float* aggr2 = (float*)(ws + o_aggr2);
    float* pool  = (float*)(ws + o_pool);
    float* cnt   = (float*)(ws + o_cnt);
    unsigned short* ea    = (unsigned short*)(ws + o_ea);
    unsigned short* eat   = (unsigned short*)(ws + o_eat);
    unsigned short* t1    = (unsigned short*)(ws + o_t1);
    float*          h1    = (float*)(ws + o_h1);
    unsigned short* t2    = (unsigned short*)(ws + o_t2);
    unsigned short* ew1t  = (unsigned short*)(ws + o_ew1t);
    unsigned short* ew2t  = (unsigned short*)(ws + o_ew2t);
    unsigned short* le1t  = (unsigned short*)(ws + o_le1t);
    unsigned short* le2t  = (unsigned short*)(ws + o_le2t);
    unsigned short* n1w1t = (unsigned short*)(ws + o_n1w1t);
    unsigned short* n1w2t = (unsigned short*)(ws + o_n1w2t);
    unsigned short* n2w1t = (unsigned short*)(ws + o_n2w1t);
    unsigned short* n2w2t = (unsigned short*)(ws + o_n2w2t);

    auto gsz = [](size_t n, int b) { return (int)((n + b - 1) / b); };

    // 0) zero accumulators
    k_zero_f32<<<gsz(zero_floats, 256), 256, 0, stream>>>((float*)ws, (int)zero_floats);

    // 1) convert edge_attr, transpose+convert all GEMM weights to bf16
    k_f32_to_bf16<<<gsz((size_t)NE * EIN, 256), 256, 0, stream>>>(edge_attr, eat, NE * EIN);
    k_wt_bf16<<<gsz((size_t)HD * 32, 256), 256, 0, stream>>>(ew1, ew1t, EIN, HD, 32);
    k_wt_bf16<<<gsz((size_t)HD * HD, 256), 256, 0, stream>>>(ew2, ew2t, HD, HD, HD);
    k_wt_bf16<<<gsz((size_t)DIN * HD, 256), 256, 0, stream>>>(le1w, le1t, HD, DIN, HD);
    k_wt_bf16<<<gsz((size_t)HD * HD, 256), 256, 0, stream>>>(le2w, le2t, HD, HD, HD);
    k_wt_bf16<<<gsz((size_t)HD * DIN, 256), 256, 0, stream>>>(n1w1, n1w1t, DIN, HD, DIN);
    k_wt_bf16<<<gsz((size_t)HD * HD, 256), 256, 0, stream>>>(n1w2, n1w2t, HD, HD, HD);
    k_wt_bf16<<<gsz((size_t)HD * HD, 256), 256, 0, stream>>>(n2w1, n2w1t, HD, HD, HD);
    k_wt_bf16<<<gsz((size_t)HD * HD, 256), 256, 0, stream>>>(n2w2, n2w2t, HD, HD, HD);

    // 2) edge MLP -> ea bf16 [E,256]
    k_edge_mlp<<<NE / 64, 256, 0, stream>>>(eat, ew1t, eb1, ew2t, eb2, ea);

    // 3) conv1: fused projection (256->64) + message + scatter
    k_conv_msg<DIN><<<NE / 16 / 8, 256, 0, stream>>>(ea, le1t, le1b, x, srcI, dstI, aggr1);
    k_add_bf16<<<gsz((size_t)NN * DIN, 256), 256, 0, stream>>>(x, aggr1, t1, NN * DIN);
    k_node_mlp<DIN, true, false><<<gsz(NN, 64), 256, 0, stream>>>(
        t1, n1w1t, n1b1, n1w2t, n1b2, h1, nullptr, nullptr);

    // 4) conv2: fused projection (256->256) + message + scatter
    k_conv_msg<HD><<<NE / 16 / 8, 256, 0, stream>>>(ea, le2t, le2b, h1, srcI, dstI, aggr2);
    k_add_bf16<<<gsz((size_t)NN * HD, 256), 256, 0, stream>>>(h1, aggr2, t2, NN * HD);
    k_node_mlp<HD, false, true><<<gsz(NN, 64), 256, 0, stream>>>(
        t2, n2w1t, n2b1, n2w2t, n2b2, nullptr, batch, pool);

    // 5) mean-pool counts + readout heads
    k_counts<<<gsz(NN, 256), 256, 0, stream>>>(batch, cnt);
    k_readout<<<1, 256, 0, stream>>>(pool, cnt, l1w, l1b, l2w, l2b,
                                     hsw, hsb, hpw, hpb, hnw, hnb, (float*)d_out);
}
